// WanSelfAttention_13460427506178
// MI455X (gfx1250) — matrix operand order repaired and checked by reading, not compiled, not run on hardware
//
#include <hip/hip_runtime.h>
#include <hip/hip_bf16.h>

// ---------------- problem constants (match reference) ----------------
#define DIM    1536
#define HEADS  12
#define HD     128            // DIM / HEADS
#define CC     64             // HD/2 rope pairs per head
#define C0C    22             // C - 2*(C//3)
#define C1C    21             // C//3
#define SEQ    4356           // 9*22*22
#define SPAD   4416           // SEQ padded to multiple of 64 (69*64)
#define F_DIM  9
#define H_DIM  22
#define W_DIM  22
#define XS     484            // H_*W_ (ref keys)
#define NCLASS 2
#define EPSV   1e-6f
#define SCALE_ATT 0.08838834764831845f   // 1/sqrt(128)

typedef __attribute__((ext_vector_type(16))) __bf16 v16bf;
typedef __attribute__((ext_vector_type(8)))  __bf16 v8bf;
typedef __attribute__((ext_vector_type(8)))  float  v8f;

__device__ __forceinline__ v8f wmma_bf16(const v16bf& a, const v16bf& b, const v8f& c) {
  return __builtin_amdgcn_wmma_f32_16x16x32_bf16(false, a, false, b, (short)0, c, false, false);
}

__device__ __forceinline__ v8f zero8() {
  v8f z;
  #pragma unroll
  for (int i = 0; i < 8; ++i) z[i] = 0.f;
  return z;
}

// ---------------------------------------------------------------------
// fp32 -> bf16 with row zero-padding (rows..rowsPad zero-filled)
// ---------------------------------------------------------------------
__global__ void convert_pad(const float* __restrict__ src, __bf16* __restrict__ dst,
                            int rows, int cols, int rowsPad) {
  const size_t total = (size_t)rowsPad * (size_t)cols;
  for (size_t i = (size_t)blockIdx.x * blockDim.x + threadIdx.x; i < total;
       i += (size_t)gridDim.x * blockDim.x) {
    const size_t r = i / (size_t)cols;
    dst[i] = (r < (size_t)rows) ? (__bf16)src[i] : (__bf16)0.f;
  }
}

__global__ void fill_zero_bf16(__bf16* __restrict__ p, size_t n) {
  for (size_t i = (size_t)blockIdx.x * blockDim.x + threadIdx.x; i < n;
       i += (size_t)gridDim.x * blockDim.x)
    p[i] = (__bf16)0.f;
}

// ---------------------------------------------------------------------
// C[M x N] = A[M x K] @ B[N x K]^T + bias   (A,B bf16, C fp32)
// BM=64 BN=128 BK=32, 256 threads = 8 waves, each wave a 32x32 tile.
// WMMA 16x16x32 bf16, A and B fragments in contiguous k order (k = 16*half + s).
// ---------------------------------------------------------------------
__global__ __launch_bounds__(256) void gemm_bias(
    const __bf16* __restrict__ A, const __bf16* __restrict__ B,
    const float* __restrict__ bias, float* __restrict__ C,
    int Mvalid, int N, int K) {
  const int tid  = threadIdx.x;
  const int lane = tid & 31;
  const int wave = tid >> 5;
  const int wrow = (wave >> 2) * 32;   // 0 / 32
  const int wcol = (wave & 3) * 32;    // 0..96
  const int l16  = lane & 15;
  const int kh   = lane >> 4;
  const int m0   = blockIdx.x * 64;
  const int n0   = blockIdx.y * 128;

  __shared__ __attribute__((aligned(64))) __bf16 As[64][32];
  __shared__ __attribute__((aligned(64))) __bf16 Bs[128][32];

  v8f acc[2][2];
  #pragma unroll
  for (int i = 0; i < 2; ++i)
    #pragma unroll
    for (int j = 0; j < 2; ++j) acc[i][j] = zero8();

  const int arow = tid >> 2, acol = (tid & 3) * 8;   // 64x32 tile, 8 bf16/thread
  const int brow = tid >> 1, bcol = (tid & 1) * 16;  // 128x32 tile, 16 bf16/thread

  for (int k0 = 0; k0 < K; k0 += 32) {
    __syncthreads();
    *(v8bf*)&As[arow][acol]     = *(const v8bf*)&A[(size_t)(m0 + arow) * K + k0 + acol];
    *(v8bf*)&Bs[brow][bcol]     = *(const v8bf*)&B[(size_t)(n0 + brow) * K + k0 + bcol];
    *(v8bf*)&Bs[brow][bcol + 8] = *(const v8bf*)&B[(size_t)(n0 + brow) * K + k0 + bcol + 8];
    if (k0 + 32 < K) {  // global_prefetch_b8 of next K-slab
      __builtin_prefetch(&A[(size_t)(m0 + arow) * K + k0 + 32 + acol], 0, 1);
      __builtin_prefetch(&B[(size_t)(n0 + brow) * K + k0 + 32 + bcol], 0, 1);
    }
    __syncthreads();

    v16bf af[2], bf[2];
    #pragma unroll
    for (int i = 0; i < 2; ++i) {             // A frag: 16x32, lane m=l16, K halves
      const int m = wrow + i * 16 + l16;
      #pragma unroll
      for (int e = 0; e < 8; ++e) {
        af[i][e]     = As[m][kh * 16 + e];
        af[i][e + 8] = As[m][kh * 16 + 8 + e];
      }
    }
    #pragma unroll
    for (int j = 0; j < 2; ++j) {             // B frag: lanes0-15 K=0..15 contiguous
      const int n = wcol + j * 16 + l16;
      bf[j] = *(const v16bf*)&Bs[n][kh * 16];
    }
    #pragma unroll
    for (int i = 0; i < 2; ++i)
      #pragma unroll
      for (int j = 0; j < 2; ++j)
        acc[i][j] = wmma_bf16(af[i], bf[j], acc[i][j]);
  }

  #pragma unroll
  for (int i = 0; i < 2; ++i)
    #pragma unroll
    for (int j = 0; j < 2; ++j) {
      const int n  = n0 + wcol + j * 16 + l16;
      const float bv = bias[n];
      #pragma unroll
      for (int r = 0; r < 8; ++r) {
        const int m = m0 + wrow + i * 16 + kh * 8 + r;  // C layout: M = lg*8 + r
        if (m < Mvalid) C[(size_t)m * N + n] = acc[i][j][r] + bv;
      }
    }
}

// ---------------------------------------------------------------------
// rmsnorm over full row (DIM) * g, then RoPE per head with grid cos/sin.
// Output bf16 head-major [HEADS][SPAD][HD]. One block per sequence pos.
// ---------------------------------------------------------------------
__global__ __launch_bounds__(256) void rmsnorm_rope(
    const float* __restrict__ Y, const float* __restrict__ g,
    const float* __restrict__ fc, const float* __restrict__ fs,
    __bf16* __restrict__ outb, int SP) {
  const int s = blockIdx.x;
  const float* row = Y + (size_t)s * DIM;
  __shared__ float red[256];
  float acc = 0.f;
  for (int i = threadIdx.x; i < DIM; i += 256) { float v = row[i]; acc += v * v; }
  red[threadIdx.x] = acc;
  __syncthreads();
  for (int st = 128; st > 0; st >>= 1) {
    if (threadIdx.x < st) red[threadIdx.x] += red[threadIdx.x + st];
    __syncthreads();
  }
  const float rms = rsqrtf(red[0] * (1.f / DIM) + EPSV);

  const int hw  = H_DIM * W_DIM;
  const int fi  = s / hw;
  const int rem = s - fi * hw;
  const int hi  = rem / W_DIM;
  const int wi  = rem - hi * W_DIM;

  for (int p = threadIdx.x; p < HEADS * CC; p += 256) {
    const int head = p >> 6;            // p / 64
    const int c    = p & 63;            // pair index within head
    const int pos  = (c < C0C) ? fi : ((c < C0C + C1C) ? hi : wi);
    const float co = fc[pos * CC + c];
    const float si = fs[pos * CC + c];
    const int idx  = head * HD + 2 * c;
    const float a = row[idx]     * rms * g[idx];
    const float b = row[idx + 1] * rms * g[idx + 1];
    __bf16* o = outb + ((size_t)head * SP + s) * HD + 2 * c;
    o[0] = (__bf16)(a * co - b * si);
    o[1] = (__bf16)(a * si + b * co);
  }
}

// V: fp32 [S][DIM] -> bf16 head-major [HEADS][SPAD][HD] with zero pad
__global__ void pack_heads(const float* __restrict__ V, __bf16* __restrict__ Vb,
                           int S, int SP) {
  const size_t total = (size_t)HEADS * SP * HD;
  for (size_t i = (size_t)blockIdx.x * blockDim.x + threadIdx.x; i < total;
       i += (size_t)gridDim.x * blockDim.x) {
    const int dd = (int)(i % HD);
    const size_t t = i / HD;
    const int s = (int)(t % SP);
    const int head = (int)(t / SP);
    Vb[i] = (s < S) ? (__bf16)V[(size_t)s * DIM + head * HD + dd] : (__bf16)0.f;
  }
}

// ---------------------------------------------------------------------
// Flash attention, one head per blockIdx.y, 64 query rows per block
// (4 waves x 16 rows). Key tiles of 32, online softmax, WMMA for both
// Q@K^T and P@V. O written fp32 [s][DIM] (head columns).
// ---------------------------------------------------------------------
__global__ __launch_bounds__(128) void flash_attn(
    const __bf16* __restrict__ Qb, const __bf16* __restrict__ Kb,
    const __bf16* __restrict__ Vb, float* __restrict__ O, int S, int SP) {
  const int head = blockIdx.y;
  const int tid  = threadIdx.x;
  const int lane = tid & 31;
  const int wave = tid >> 5;
  const int l16  = lane & 15;
  const int kh   = lane >> 4;
  const int qrow0 = blockIdx.x * 64 + wave * 16;

  __shared__ __attribute__((aligned(64))) __bf16 Ks[32][HD];    // [key][d]
  __shared__ __attribute__((aligned(64))) __bf16 Vts[HD][32];   // [d][key] transposed
  __shared__ __attribute__((aligned(64))) __bf16 Ps[4][16][32]; // per-wave P staging

  const __bf16* Qh = Qb + (size_t)head * SP * HD;
  const __bf16* Kh = Kb + (size_t)head * SP * HD;
  const __bf16* Vh = Vb + (size_t)head * SP * HD;

  // Q fragments (A: 16x32 bf16) for the 4 K-chunks of d=128
  v16bf qf[4];
  {
    const __bf16* qrow = Qh + (size_t)(qrow0 + l16) * HD;
    #pragma unroll
    for (int kk = 0; kk < 4; ++kk)
      #pragma unroll
      for (int e = 0; e < 8; ++e) {
        qf[kk][e]     = qrow[kk * 32 + kh * 16 + e];
        qf[kk][e + 8] = qrow[kk * 32 + kh * 16 + 8 + e];
      }
  }

  float mrun[8], lrun[8];
  v8f oacc[8];
  #pragma unroll
  for (int r = 0; r < 8; ++r) { mrun[r] = -1e30f; lrun[r] = 0.f; }
  #pragma unroll
  for (int d = 0; d < 8; ++d) oacc[d] = zero8();

  const int krow = tid >> 2;         // 0..31
  const int kcol = (tid & 3) * 32;   // 0,32,64,96
  const int ntiles = SP / 32;

  for (int t = 0; t < ntiles; ++t) {
    __syncthreads();
    {
      const __bf16* ksrc = Kh + (size_t)(t * 32 + krow) * HD + kcol;
      *(v8bf*)&Ks[krow][kcol]      = *(const v8bf*)&ksrc[0];
      *(v8bf*)&Ks[krow][kcol + 8]  = *(const v8bf*)&ksrc[8];
      *(v8bf*)&Ks[krow][kcol + 16] = *(const v8bf*)&ksrc[16];
      *(v8bf*)&Ks[krow][kcol + 24] = *(const v8bf*)&ksrc[24];
      const __bf16* vsrc = Vh + (size_t)(t * 32 + krow) * HD + kcol;
      #pragma unroll
      for (int e = 0; e < 32; ++e) Vts[kcol + e][krow] = vsrc[e];  // transpose into LDS
    }
    __syncthreads();

    // scores: Q(16x128) @ K^T -> two 16x16 tiles (keys 0..15, 16..31)
    v8f scv[2];
    scv[0] = zero8(); scv[1] = zero8();
    #pragma unroll
    for (int hhalf = 0; hhalf < 2; ++hhalf) {
      const int n = hhalf * 16 + l16;  // key-in-tile (B-frag lane)
      #pragma unroll
      for (int kk = 0; kk < 4; ++kk) {
        v16bf bfr = *(const v16bf*)&Ks[n][kk * 32 + kh * 16];
        scv[hhalf] = wmma_bf16(qf[kk], bfr, scv[hhalf]);
      }
    }
    // scale + mask padded keys
    #pragma unroll
    for (int hhalf = 0; hhalf < 2; ++hhalf) {
      const int key = t * 32 + hhalf * 16 + l16;
      const bool valid = key < S;
      #pragma unroll
      for (int r = 0; r < 8; ++r)
        scv[hhalf][r] = valid ? scv[hhalf][r] * SCALE_ATT : -1e30f;
    }
    // online softmax per row (rows of this lane group: m = kh*8 + r)
    #pragma unroll
    for (int r = 0; r < 8; ++r) {
      float mx = fmaxf(scv[0][r], scv[1][r]);
      mx = fmaxf(mx, __shfl_xor(mx, 1, 32));
      mx = fmaxf(mx, __shfl_xor(mx, 2, 32));
      mx = fmaxf(mx, __shfl_xor(mx, 4, 32));
      mx = fmaxf(mx, __shfl_xor(mx, 8, 32));
      const float mnew  = fmaxf(mrun[r], mx);
      const float alpha = __expf(mrun[r] - mnew);
      mrun[r] = mnew;
      const float p0 = __expf(scv[0][r] - mnew);
      const float p1 = __expf(scv[1][r] - mnew);
      scv[0][r] = p0; scv[1][r] = p1;
      float ps = p0 + p1;
      ps += __shfl_xor(ps, 1, 32);
      ps += __shfl_xor(ps, 2, 32);
      ps += __shfl_xor(ps, 4, 32);
      ps += __shfl_xor(ps, 8, 32);
      lrun[r] = lrun[r] * alpha + ps;
      #pragma unroll
      for (int d = 0; d < 8; ++d) oacc[d][r] *= alpha;
    }
    // stage P (C layout) -> LDS -> reload as A fragment (per-wave buffer,
    // same-wave LDS ops are in-order)
    #pragma unroll
    for (int hhalf = 0; hhalf < 2; ++hhalf)
      #pragma unroll
      for (int r = 0; r < 8; ++r)
        Ps[wave][kh * 8 + r][hhalf * 16 + l16] = (__bf16)scv[hhalf][r];
    __builtin_amdgcn_wave_barrier();
    v16bf pf;
    #pragma unroll
    for (int e = 0; e < 8; ++e) {
      pf[e]     = Ps[wave][l16][kh * 16 + e];
      pf[e + 8] = Ps[wave][l16][kh * 16 + 8 + e];
    }
    // P(16x32) @ V(32x128): 8 WMMA over the d tiles, B-frags from Vts rows
    #pragma unroll
    for (int d = 0; d < 8; ++d) {
      v16bf vfr = *(const v16bf*)&Vts[d * 16 + l16][kh * 16];
      oacc[d] = wmma_bf16(pf, vfr, oacc[d]);
    }
  }

  #pragma unroll
  for (int d = 0; d < 8; ++d) {
    const int n = head * HD + d * 16 + l16;
    #pragma unroll
    for (int r = 0; r < 8; ++r) {
      const int m = qrow0 + kh * 8 + r;
      if (m < S) O[(size_t)m * DIM + n] = oacc[d][r] / lrun[r];
    }
  }
}

// ---------------------------------------------------------------------
// attn_map: softmax(q*scale @ k[:484]^T) dotted with masks, mean over
// heads, / mask-sum. One wave per 16 query rows; WMMA for scores.
// amap layout: [NCLASS][SEQ]
// ---------------------------------------------------------------------
__global__ __launch_bounds__(32) void attn_map_kernel(
    const __bf16* __restrict__ Qb, const __bf16* __restrict__ Kb,
    const float* __restrict__ masks, float* __restrict__ amap, int S, int SP) {
  const int lane = threadIdx.x;
  const int l16  = lane & 15;
  const int half = lane >> 4;
  const int qrow0 = blockIdx.x * 16;
  __shared__ __attribute__((aligned(64))) float sc[16][496];

  float ms0 = 0.f, ms1 = 0.f;
  for (int t = lane; t < XS; t += 32) { ms0 += masks[t]; ms1 += masks[XS + t]; }
  #pragma unroll
  for (int off = 16; off > 0; off >>= 1) {
    ms0 += __shfl_xor(ms0, off, 32);
    ms1 += __shfl_xor(ms1, off, 32);
  }

  float row0 = 0.f, row1 = 0.f;
  for (int head = 0; head < HEADS; ++head) {
    const __bf16* Qh = Qb + (size_t)head * SP * HD;
    const __bf16* Kh = Kb + (size_t)head * SP * HD;
    v16bf qf[4];
    {
      const __bf16* qrow = Qh + (size_t)(qrow0 + l16) * HD;
      #pragma unroll
      for (int kk = 0; kk < 4; ++kk)
        #pragma unroll
        for (int e = 0; e < 8; ++e) {
          qf[kk][e]     = qrow[kk * 32 + half * 16 + e];
          qf[kk][e + 8] = qrow[kk * 32 + half * 16 + 8 + e];
        }
    }
    __syncthreads();
    for (int kt = 0; kt < 31; ++kt) {       // 31*16 = 496 >= 484 keys
      v8f c = zero8();
      #pragma unroll
      for (int kk = 0; kk < 4; ++kk) {
        v16bf bfr = *(const v16bf*)&Kh[(size_t)(kt * 16 + l16) * HD + kk * 32 + half * 16];
        c = wmma_bf16(qf[kk], bfr, c);
      }
      const int col = kt * 16 + l16;
      #pragma unroll
      for (int r = 0; r < 8; ++r) sc[half * 8 + r][col] = c[r] * SCALE_ATT;
    }
    __syncthreads();
    // softmax over t<484 + mask dots, 2 lanes per row
    const int m = l16;
    float mx = -1e30f;
    for (int t = half; t < XS; t += 2) mx = fmaxf(mx, sc[m][t]);
    mx = fmaxf(mx, __shfl_xor(mx, 16, 32));
    float se = 0.f, d0 = 0.f, d1 = 0.f;
    for (int t = half; t < XS; t += 2) {
      const float p = __expf(sc[m][t] - mx);
      se += p; d0 += p * masks[t]; d1 += p * masks[XS + t];
    }
    se += __shfl_xor(se, 16, 32);
    d0 += __shfl_xor(d0, 16, 32);
    d1 += __shfl_xor(d1, 16, 32);
    row0 += d0 / se;
    row1 += d1 / se;
  }
  const int s = qrow0 + l16;
  if (half == 0 && s < S) {
    amap[s]     = row0 / ((float)HEADS * ms0);
    amap[S + s] = row1 / ((float)HEADS * ms1);
  }
}

// ---------------------------------------------------------------------
extern "C" void kernel_launch(void* const* d_in, const int* in_sizes, int n_in,
                              void* d_out, int out_size, void* d_ws, size_t ws_size,
                              hipStream_t stream) {
  (void)in_sizes; (void)n_in; (void)out_size; (void)ws_size;
  const float* x   = (const float*)d_in[0];
  const float* Wq  = (const float*)d_in[1];
  const float* bq  = (const float*)d_in[2];
  const float* Wk  = (const float*)d_in[3];
  const float* bk  = (const float*)d_in[4];
  const float* Wv  = (const float*)d_in[5];
  const float* bv  = (const float*)d_in[6];
  const float* Wo  = (const float*)d_in[7];
  const float* bo  = (const float*)d_in[8];
  const float* gq  = (const float*)d_in[9];
  const float* gk  = (const float*)d_in[10];
  const float* fc  = (const float*)d_in[11];
  const float* fs  = (const float*)d_in[12];
  const float* msk = (const float*)d_in[13];
  float* out = (float*)d_out;

  // workspace carving (raw fp32 buffer reused for Qraw/Kraw/Vraw and O)
  char* ws = (char*)d_ws;
  size_t off = 0;
  auto carve = [&](size_t bytes) -> void* {
    void* p = ws + off;
    off += (bytes + 255) & ~(size_t)255;
    return p;
  };
  __bf16* Xb  = (__bf16*)carve((size_t)SPAD * DIM * 2);
  __bf16* Wqb = (__bf16*)carve((size_t)DIM * DIM * 2);
  __bf16* Wkb = (__bf16*)carve((size_t)DIM * DIM * 2);
  __bf16* Wvb = (__bf16*)carve((size_t)DIM * DIM * 2);
  __bf16* Wob = (__bf16*)carve((size_t)DIM * DIM * 2);
  float*  raw = (float*)carve((size_t)SEQ * DIM * 4);
  __bf16* Qb  = (__bf16*)carve((size_t)HEADS * SPAD * HD * 2);
  __bf16* Kb  = (__bf16*)carve((size_t)HEADS * SPAD * HD * 2);
  __bf16* Vb  = (__bf16*)carve((size_t)HEADS * SPAD * HD * 2);
  __bf16* Ob  = Xb;  // Xb dead after the three input GEMMs

  const dim3 gGemm(SPAD / 64, DIM / 128);

  convert_pad<<<2048, 256, 0, stream>>>(x,  Xb,  SEQ, DIM, SPAD);
  convert_pad<<<2048, 256, 0, stream>>>(Wq, Wqb, DIM, DIM, DIM);
  convert_pad<<<2048, 256, 0, stream>>>(Wk, Wkb, DIM, DIM, DIM);
  convert_pad<<<2048, 256, 0, stream>>>(Wv, Wvb, DIM, DIM, DIM);
  convert_pad<<<2048, 256, 0, stream>>>(Wo, Wob, DIM, DIM, DIM);
  fill_zero_bf16<<<2048, 256, 0, stream>>>(Qb, (size_t)HEADS * SPAD * HD);
  fill_zero_bf16<<<2048, 256, 0, stream>>>(Kb, (size_t)HEADS * SPAD * HD);

  // Q projection -> rmsnorm+rope -> Qb
  gemm_bias<<<gGemm, 256, 0, stream>>>(Xb, Wqb, bq, raw, SEQ, DIM, DIM);
  rmsnorm_rope<<<SEQ, 256, 0, stream>>>(raw, gq, fc, fs, Qb, SPAD);
  // K projection -> rmsnorm+rope -> Kb
  gemm_bias<<<gGemm, 256, 0, stream>>>(Xb, Wkb, bk, raw, SEQ, DIM, DIM);
  rmsnorm_rope<<<SEQ, 256, 0, stream>>>(raw, gk, fc, fs, Kb, SPAD);
  // V projection -> head-major bf16
  gemm_bias<<<gGemm, 256, 0, stream>>>(Xb, Wvb, bv, raw, SEQ, DIM, DIM);
  pack_heads<<<2048, 256, 0, stream>>>(raw, Vb, SEQ, SPAD);

  // flash attention (O reuses raw)
  flash_attn<<<dim3(SPAD / 64, HEADS), 128, 0, stream>>>(Qb, Kb, Vb, raw, SEQ, SPAD);

  // output projection: O -> bf16 (reusing Xb) -> out = Ob @ Wo^T + bo
  convert_pad<<<2048, 256, 0, stream>>>(raw, Ob, SEQ, DIM, SPAD);
  gemm_bias<<<gGemm, 256, 0, stream>>>(Ob, Wob, bo, out, SEQ, DIM, DIM);

  // auxiliary attention map into out[SEQ*DIM ..]
  attn_map_kernel<<<(SEQ + 15) / 16, 32, 0, stream>>>(
      Qb, Kb, msk, out + (size_t)SEQ * DIM, SEQ, SPAD);
}
